// EfficientLoFTRAggregatedAttention_64544768524413
// MI455X (gfx1250) — matrix operand order repaired and checked by reading, not compiled, not run on hardware
//
#include <hip/hip_runtime.h>
#include <hip/hip_bf16.h>
#include <math.h>

typedef __attribute__((ext_vector_type(16))) _Float16 v16h;
typedef __attribute__((ext_vector_type(8)))  _Float16 v8h;
typedef __attribute__((ext_vector_type(8)))  float    v8f;

#define SEQ   1600
#define CDIM  256
#define HDIM  160
#define HWSZ  25600
#define INTER 1024

// ---------------------------------------------------------------------------
// WMMA fragment helpers (CDNA5 wave32 layouts, ISA 7.12.2)
// A 16x32 f16: lane m=lane&15, hi=lane>>4: elems 0..7 -> k=hi*8+e, 8..15 -> k=16+hi*8+(e-8)
// B 32x16 f16 from a transposed (NxK) store: elems 0..7 -> k=hi*8+e, 8..15 -> k=16+hi*8+(e-8), col = lane&15
// C/D 16x16 f32: vgpr r -> row r + hi*8, col = lane&15
// ---------------------------------------------------------------------------
static __device__ __forceinline__ v16h frag_a(const _Float16* p, int ld) {
  int lane = threadIdx.x & 31;
  int m = lane & 15, hi = lane >> 4;
  const _Float16* q = p + m * ld + hi * 8;
  union { v16h v; v8h h[2]; } u;
  u.h[0] = *(const v8h*)q;
  u.h[1] = *(const v8h*)(q + 16);
  return u.v;
}

static __device__ __forceinline__ v16h frag_bt(const _Float16* p, int ld) {
  int lane = threadIdx.x & 31;
  int n = lane & 15, hi = lane >> 4;
  const _Float16* q = p + n * ld + hi * 8;
  union { v16h v; v8h h[2]; } u;
  u.h[0] = *(const v8h*)q;
  u.h[1] = *(const v8h*)(q + 16);
  return u.v;
}

static __device__ __forceinline__ v8f wmma16(v16h a, v16h b, v8f c) {
  return __builtin_amdgcn_wmma_f32_16x16x32_f16(false, a, false, b, (short)0, c,
                                                false, false);
}

// ---------------------------------------------------------------------------
// Weight convert + transpose: (K,N) f32 -> (N,K) f16
// ---------------------------------------------------------------------------
__global__ __launch_bounds__(256) void k_wt(const float* __restrict__ w,
                                            _Float16* __restrict__ wt,
                                            int K, int N) {
  size_t i = (size_t)blockIdx.x * 256 + threadIdx.x;
  if (i < (size_t)K * N) {
    int k = (int)(i / N), n = (int)(i % N);
    wt[(size_t)n * K + k] = (_Float16)w[i];
  }
}

// ---------------------------------------------------------------------------
// Aggregation (depthwise 4x4 conv + 4x4 maxpool) + channel LayerNorm -> f16
// block = (b, s) token, 256 threads = channels
// ---------------------------------------------------------------------------
__global__ __launch_bounds__(256) void k_agg(
    const float* __restrict__ hs, const float* __restrict__ es,
    const float* __restrict__ qw, const float* __restrict__ gg,
    const float* __restrict__ gb,
    _Float16* __restrict__ QX, _Float16* __restrict__ KVX) {
  __shared__ float red[256], red2[256];
  int b = blockIdx.x / SEQ, s = blockIdx.x % SEQ;
  int ay = s / 40, ax = s % 40;
  int c = threadIdx.x;
  const float* hp = hs + ((size_t)b * CDIM + c) * HWSZ;
  const float* ep = es + ((size_t)b * CDIM + c) * HWSZ;
  float acc = 0.f, mx = -3.0e38f;
#pragma unroll
  for (int i = 0; i < 4; ++i) {
#pragma unroll
    for (int j = 0; j < 4; ++j) {
      int off = (ay * 4 + i) * HDIM + ax * 4 + j;
      acc = fmaf(hp[off], qw[c * 16 + i * 4 + j], acc);
      mx = fmaxf(mx, ep[off]);
    }
  }
  float vals[2] = {acc, mx};
  _Float16* outs[2] = {QX, KVX};
  for (int t = 0; t < 2; ++t) {
    float v = vals[t];
    red[c] = v; red2[c] = v * v;
    __syncthreads();
    for (int off = 128; off > 0; off >>= 1) {
      if (c < off) { red[c] += red[c + off]; red2[c] += red2[c + off]; }
      __syncthreads();
    }
    float mu  = red[0]  * (1.f / 256.f);
    float var = red2[0] * (1.f / 256.f) - mu * mu;
    float o = (v - mu) * rsqrtf(var + 1e-5f) * gg[c] + gb[c];
    outs[t][(size_t)blockIdx.x * CDIM + c] = (_Float16)o;
    __syncthreads();
  }
}

// ---------------------------------------------------------------------------
// Generic GEMM: C(MxN,f32) = A(MxK,f16 row-major) * B (given as BT: NxK f16)
// block tile 64x64, 8 waves, k-chunk 32, A staged in LDS
// ---------------------------------------------------------------------------
__global__ __launch_bounds__(256) void k_gemm(
    const _Float16* __restrict__ A, const _Float16* __restrict__ BT,
    float* __restrict__ Cg, int M, int N, int K) {
  __shared__ _Float16 As[64 * 32];
  int tilesN = N >> 6;
  int bm = blockIdx.x / tilesN, bn = blockIdx.x % tilesN;
  int m0 = bm << 6, n0 = bn << 6;
  int tid = threadIdx.x, lane = tid & 31, wv = tid >> 5;
  int mrow = wv >> 1;                          // 0..3
  int ncol0 = n0 + ((wv & 1) * 2) * 16;        // two adjacent 16-wide n tiles
  v8f acc0 = {}, acc1 = {};
  int flat = tid * 8;
  int ar = flat >> 5, ac = flat & 31;
  for (int k0 = 0; k0 < K; k0 += 32) {
    __syncthreads();
    *(v8h*)(As + flat) = *(const v8h*)(A + (size_t)(m0 + ar) * K + k0 + ac);
    if (k0 + 32 < K)
      __builtin_prefetch((const void*)(A + (size_t)(m0 + ar) * K + k0 + 32 + ac), 0, 1);
    __syncthreads();
    v16h af = frag_a(As + mrow * 16 * 32, 32);
    v16h b0 = frag_bt(BT + (size_t)ncol0 * K + k0, K);
    v16h b1 = frag_bt(BT + (size_t)(ncol0 + 16) * K + k0, K);
    acc0 = wmma16(af, b0, acc0);
    acc1 = wmma16(af, b1, acc1);
  }
  int hi = lane >> 4, n = lane & 15;
  int row = m0 + mrow * 16 + hi * 8;
#pragma unroll
  for (int r = 0; r < 8; ++r) {
    Cg[(size_t)(row + r) * N + ncol0 + n]      = acc0[r];
    Cg[(size_t)(row + r) * N + ncol0 + 16 + n] = acc1[r];
  }
}

// ---------------------------------------------------------------------------
// RoPE on Q,K (full C) + head split; V copied transposed (b,h,d,s) for B-frags
// ---------------------------------------------------------------------------
__global__ __launch_bounds__(256) void k_rope(
    const float* __restrict__ qf, const float* __restrict__ kf,
    const float* __restrict__ vf, const float* __restrict__ cosp,
    const float* __restrict__ sinp, _Float16* __restrict__ Qh,
    _Float16* __restrict__ Kh, _Float16* __restrict__ VT) {
  int s = blockIdx.x % SEQ, b = blockIdx.x / SEQ;
  int c = threadIdx.x;
  size_t idx = (size_t)blockIdx.x * CDIM + c;
  float cv = cosp[idx], sv = sinp[idx];
  float qo, ko;
  if ((c & 1) == 0) {  // out[2i] = x[2i]*cos - x[2i+1]*sin
    qo = qf[idx] * cv - qf[idx + 1] * sv;
    ko = kf[idx] * cv - kf[idx + 1] * sv;
  } else {             // out[2i+1] = x[2i+1]*cos + x[2i]*sin
    qo = qf[idx] * cv + qf[idx - 1] * sv;
    ko = kf[idx] * cv + kf[idx - 1] * sv;
  }
  int h = c >> 5, d = c & 31;
  size_t hidx = (((size_t)b * 8 + h) * SEQ + s) * 32 + d;
  Qh[hidx] = (_Float16)qo;
  Kh[hidx] = (_Float16)ko;
  VT[(((size_t)b * 8 + h) * 32 + d) * SEQ + s] = (_Float16)vf[idx];
}

// ---------------------------------------------------------------------------
// Flash attention: 1 wave per (b,h,16-row q-tile); K-dim 32 == Dh, so one WMMA
// per 16x16 score tile; P transposed via LDS into A-fragment layout.
// ---------------------------------------------------------------------------
__global__ __launch_bounds__(32) void k_attn(
    const _Float16* __restrict__ Q, const _Float16* __restrict__ Km,
    const _Float16* __restrict__ VT, _Float16* __restrict__ AO) {
  __shared__ _Float16 P[16 * 32];
  int bh = blockIdx.x / 100;       // b*8 + h
  int q0 = (blockIdx.x % 100) * 16;
  const _Float16* qp = Q  + ((size_t)bh * SEQ + q0) * 32;
  const _Float16* kp = Km + (size_t)bh * SEQ * 32;
  const _Float16* vp = VT + (size_t)bh * 32 * SEQ;
  int lane = threadIdx.x & 31, n = lane & 15, hi = lane >> 4;
  v16h qa = frag_a(qp, 32);
  v8f o0 = {}, o1 = {};
  float mrow[8], lrow[8];
#pragma unroll
  for (int r = 0; r < 8; ++r) { mrow[r] = -3.0e38f; lrow[r] = 0.f; }
  const float scale = 0.17677669529663687f;  // 32^-0.5
  for (int kv0 = 0; kv0 < SEQ; kv0 += 32) {
    v16h b0 = frag_bt(kp + (size_t)kv0 * 32, 32);
    v16h b1 = frag_bt(kp + (size_t)(kv0 + 16) * 32, 32);
    v8f s0 = {}, s1 = {};
    s0 = wmma16(qa, b0, s0);
    s1 = wmma16(qa, b1, s1);
    float p0[8], p1[8];
#pragma unroll
    for (int r = 0; r < 8; ++r) {
      float a = s0[r] * scale, b = s1[r] * scale;
      float mx = fmaxf(a, b);
      mx = fmaxf(mx, __shfl_xor(mx, 1));
      mx = fmaxf(mx, __shfl_xor(mx, 2));
      mx = fmaxf(mx, __shfl_xor(mx, 4));
      mx = fmaxf(mx, __shfl_xor(mx, 8));
      float mn = fmaxf(mrow[r], mx);
      float alpha = __expf(mrow[r] - mn);
      mrow[r] = mn;
      p0[r] = __expf(a - mn);
      p1[r] = __expf(b - mn);
      float rs = p0[r] + p1[r];
      rs += __shfl_xor(rs, 1);
      rs += __shfl_xor(rs, 2);
      rs += __shfl_xor(rs, 4);
      rs += __shfl_xor(rs, 8);
      lrow[r] = lrow[r] * alpha + rs;
      o0[r] *= alpha;
      o1[r] *= alpha;
    }
    __syncthreads();
#pragma unroll
    for (int r = 0; r < 8; ++r) {
      int row = r + hi * 8;
      P[row * 32 + n]      = (_Float16)p0[r];
      P[row * 32 + 16 + n] = (_Float16)p1[r];
    }
    __syncthreads();
    v16h pa  = frag_a(P, 32);
    v16h bv0 = frag_bt(vp + kv0, SEQ);
    v16h bv1 = frag_bt(vp + (size_t)16 * SEQ + kv0, SEQ);
    o0 = wmma16(pa, bv0, o0);
    o1 = wmma16(pa, bv1, o1);
  }
  int h = bh & 7, b = bh >> 3;
#pragma unroll
  for (int r = 0; r < 8; ++r) {
    int row = q0 + r + hi * 8;
    float inv = 1.0f / lrow[r];
    size_t base = ((size_t)b * SEQ + row) * CDIM + h * 32;
    AO[base + n]      = (_Float16)(o0[r] * inv);
    AO[base + 16 + n] = (_Float16)(o1[r] * inv);
  }
}

// ---------------------------------------------------------------------------
// Fused bilinear-upsample + concat + FC1(leaky relu) + FC2 + LayerNorm +
// residual.  16 tokens/block, 4 waves; FC1 done in two 512-col halves so the
// h-tile LDS stays at 16 KB; FC2 accumulators persist across halves.
// ---------------------------------------------------------------------------
__global__ __launch_bounds__(128) void k_mlp(
    const float* __restrict__ hs, const float* __restrict__ aop,
    const _Float16* __restrict__ FC1T, const _Float16* __restrict__ FC2T,
    const float* __restrict__ lng, const float* __restrict__ lnb,
    float* __restrict__ out) {
  __shared__ _Float16 xs[16 * 512];    // input tile  (16 KB)
  __shared__ _Float16 hsr[16 * 512];   // half of fc1 output (16 KB)
  __shared__ float outs[16 * 256];     // fc2 output (16 KB)
  __shared__ float red[16 * 8], red2[16 * 8], mu_s[16], rs_s[16];
  int tid = threadIdx.x, lane = tid & 31, wv = tid >> 5;
  int t0 = blockIdx.x * 16;

  // ---- load x tile: [hidden | bilinear(upsample(ao_proj))] ----
  for (int i = tid; i < 16 * 512; i += 128) {
    int row = i >> 9, col = i & 511;
    int t = t0 + row;
    int b = t / HWSZ, rem = t % HWSZ;
    int y = rem / HDIM, x = rem % HDIM;
    float val;
    if (col < 256) {
      val = hs[((size_t)b * CDIM + col) * HWSZ + rem];
    } else {
      int c = col - 256;
      float fy = fminf(fmaxf((y + 0.5f) * 0.25f - 0.5f, 0.f), 39.f);
      float fx = fminf(fmaxf((x + 0.5f) * 0.25f - 0.5f, 0.f), 39.f);
      int y0 = (int)fy, x0 = (int)fx;
      int y1 = min(y0 + 1, 39), x1 = min(x0 + 1, 39);
      float wy = fy - y0, wx = fx - x0;
      const float* base = aop + (size_t)b * SEQ * CDIM;
      float v00 = base[(size_t)(y0 * 40 + x0) * CDIM + c];
      float v01 = base[(size_t)(y0 * 40 + x1) * CDIM + c];
      float v10 = base[(size_t)(y1 * 40 + x0) * CDIM + c];
      float v11 = base[(size_t)(y1 * 40 + x1) * CDIM + c];
      val = (1.f - wy) * ((1.f - wx) * v00 + wx * v01) +
            wy * ((1.f - wx) * v10 + wx * v11);
    }
    xs[i] = (_Float16)val;
  }
  __syncthreads();

  int hi = lane >> 4, n = lane & 15;
  v8f oacc[4] = {};                    // out n-tiles wv*4 .. wv*4+3
  for (int half = 0; half < 2; ++half) {
    // FC1 for cols [half*512, half*512+512): wave owns 8 n-tiles
    for (int nt = 0; nt < 8; ++nt) {
      int ncol = (wv * 8 + nt) * 16;       // col within half
      int ng = half * 512 + ncol;          // global fc1 out col
      v8f a = {};
      for (int k0 = 0; k0 < 512; k0 += 32) {
        v16h af = frag_a(xs + k0, 512);
        v16h bf = frag_bt(FC1T + (size_t)ng * 512 + k0, 512);
        a = wmma16(af, bf, a);
      }
#pragma unroll
      for (int r = 0; r < 8; ++r) {
        float v = a[r];
        v = v > 0.f ? v : 0.01f * v;       // leaky relu
        hsr[(r + hi * 8) * 512 + ncol + n] = (_Float16)v;
      }
    }
    __syncthreads();
    // FC2 partial accumulation over this K half
    for (int ot = 0; ot < 4; ++ot) {
      int ncol = (wv * 4 + ot) * 16;
      for (int k0 = 0; k0 < 512; k0 += 32) {
        v16h af = frag_a(hsr + k0, 512);
        v16h bf = frag_bt(FC2T + (size_t)ncol * INTER + half * 512 + k0, INTER);
        oacc[ot] = wmma16(af, bf, oacc[ot]);
      }
    }
    __syncthreads();
  }
  // spill fc2 result to LDS
  for (int ot = 0; ot < 4; ++ot) {
    int ncol = (wv * 4 + ot) * 16;
#pragma unroll
    for (int r = 0; r < 8; ++r)
      outs[(r + hi * 8) * 256 + ncol + n] = oacc[ot][r];
  }
  __syncthreads();
  // LayerNorm per token row (256 values)
  {
    int row = tid >> 3, sl = tid & 7;
    float sm = 0.f, sq = 0.f;
    for (int c = sl * 32; c < sl * 32 + 32; ++c) {
      float v = outs[row * 256 + c];
      sm += v; sq += v * v;
    }
    red[row * 8 + sl] = sm; red2[row * 8 + sl] = sq;
  }
  __syncthreads();
  if (tid < 16) {
    float sm = 0.f, sq = 0.f;
    for (int i = 0; i < 8; ++i) { sm += red[tid * 8 + i]; sq += red2[tid * 8 + i]; }
    float mu = sm * (1.f / 256.f);
    float var = sq * (1.f / 256.f) - mu * mu;
    mu_s[tid] = mu; rs_s[tid] = rsqrtf(var + 1e-5f);
  }
  __syncthreads();
  // residual add, NCHW store
  for (int i = tid; i < 16 * 256; i += 128) {
    int row = i >> 8, c = i & 255;
    int t = t0 + row;
    int b = t / HWSZ, rem = t % HWSZ;
    float v = (outs[i] - mu_s[row]) * rs_s[row] * lng[c] + lnb[c];
    size_t oi = ((size_t)b * CDIM + c) * HWSZ + rem;
    out[oi] = hs[oi] + v;
  }
}

// ---------------------------------------------------------------------------
extern "C" void kernel_launch(void* const* d_in, const int* in_sizes, int n_in,
                              void* d_out, int out_size, void* d_ws,
                              size_t ws_size, hipStream_t stream) {
  (void)in_sizes; (void)n_in; (void)out_size; (void)ws_size;
  const float* hs   = (const float*)d_in[0];
  const float* es   = (const float*)d_in[1];
  const float* cosp = (const float*)d_in[2];
  const float* sinp = (const float*)d_in[3];
  const float* qagg = (const float*)d_in[4];
  const float* aggg = (const float*)d_in[5];
  const float* aggb = (const float*)d_in[6];
  const float* qw   = (const float*)d_in[7];
  const float* kw   = (const float*)d_in[8];
  const float* vw   = (const float*)d_in[9];
  const float* ow   = (const float*)d_in[10];
  const float* fc1  = (const float*)d_in[11];
  const float* fc2  = (const float*)d_in[12];
  const float* lng  = (const float*)d_in[13];
  const float* lnb  = (const float*)d_in[14];
  float* out = (float*)d_out;

  char* wsp = (char*)d_ws;
  size_t off = 0;
  auto carve = [&](size_t bytes) -> void* {
    void* p = wsp + off;
    off += (bytes + 255) & ~(size_t)255;
    return p;
  };
  const size_t BS = 2 * (size_t)SEQ;  // 3200 rows
  _Float16* qx  = (_Float16*)carve(BS * CDIM * 2);
  _Float16* kvx = (_Float16*)carve(BS * CDIM * 2);
  float*    qf  = (float*)   carve(BS * CDIM * 4);
  float*    kf  = (float*)   carve(BS * CDIM * 4);
  float*    vf  = (float*)   carve(BS * CDIM * 4);
  _Float16* qh  = (_Float16*)carve(BS * CDIM * 2);
  _Float16* kh  = (_Float16*)carve(BS * CDIM * 2);
  _Float16* vT  = (_Float16*)carve(BS * CDIM * 2);
  _Float16* ao  = (_Float16*)carve(BS * CDIM * 2);
  float*    aop = (float*)   carve(BS * CDIM * 4);
  _Float16* wtq = (_Float16*)carve((size_t)CDIM * CDIM * 2);
  _Float16* wtk = (_Float16*)carve((size_t)CDIM * CDIM * 2);
  _Float16* wtv = (_Float16*)carve((size_t)CDIM * CDIM * 2);
  _Float16* wto = (_Float16*)carve((size_t)CDIM * CDIM * 2);
  _Float16* wt1 = (_Float16*)carve((size_t)512 * INTER * 2);
  _Float16* wt2 = (_Float16*)carve((size_t)INTER * CDIM * 2);

  // weight transposes (f32 -> f16 (N,K))
  k_wt<<<256, 256, 0, stream>>>(qw, wtq, CDIM, CDIM);
  k_wt<<<256, 256, 0, stream>>>(kw, wtk, CDIM, CDIM);
  k_wt<<<256, 256, 0, stream>>>(vw, wtv, CDIM, CDIM);
  k_wt<<<256, 256, 0, stream>>>(ow, wto, CDIM, CDIM);
  k_wt<<<(512 * INTER) / 256, 256, 0, stream>>>(fc1, wt1, 512, INTER);
  k_wt<<<(INTER * CDIM) / 256, 256, 0, stream>>>(fc2, wt2, INTER, CDIM);

  // aggregation + LN
  k_agg<<<(unsigned)BS, 256, 0, stream>>>(hs, es, qagg, aggg, aggb, qx, kvx);

  // QKV projections (M=3200, N=256, K=256 each)
  k_gemm<<<200, 256, 0, stream>>>(qx,  wtq, qf, 3200, 256, 256);
  k_gemm<<<200, 256, 0, stream>>>(kvx, wtk, kf, 3200, 256, 256);
  k_gemm<<<200, 256, 0, stream>>>(kvx, wtv, vf, 3200, 256, 256);

  // RoPE + head split + V transpose
  k_rope<<<(unsigned)BS, 256, 0, stream>>>(qf, kf, vf, cosp, sinp, qh, kh, vT);

  // flash attention: 2*8*100 q-tiles
  k_attn<<<1600, 32, 0, stream>>>(qh, kh, vT, ao);

  // output projection
  k_gemm<<<200, 256, 0, stream>>>(ao, wto, aop, 3200, 256, 256);

  // upsample + concat + MLP + LN + residual: 51200 tokens / 16
  k_mlp<<<3200, 128, 0, stream>>>(hs, aop, wt1, wt2, lng, lnb, out);
}
